// GAT_29231547416655
// MI455X (gfx1250) — matrix operand order repaired and checked by reading, not compiled, not run on hardware
//
#include <hip/hip_runtime.h>
#include <hip/hip_bf16.h>

#define Bsz   2
#define Nn    2048
#define FIN   128
#define FOUT  128
#define Hh    4
#define Uu    1024
#define ALPHA 0.2f

#define TILE_M 128
#define KT     32
#define APITCH 34   // pad pitch (ushorts) to break LDS bank conflicts

typedef __attribute__((ext_vector_type(16))) __bf16 v16bf;
typedef __attribute__((ext_vector_type(8)))  float  v8f;

static __device__ __forceinline__ unsigned short f2bf(float f) {
    unsigned u = __builtin_bit_cast(unsigned, f);
    unsigned r = u + 0x7FFFu + ((u >> 16) & 1u);   // round-to-nearest-even
    return (unsigned short)(r >> 16);
}
static __device__ __forceinline__ __bf16 bfc(unsigned short u) {
    return __builtin_bit_cast(__bf16, u);
}

// ---------------------------------------------------------------------------
// Kernel 1: Wh = x @ W[h]  (store bf16), s_src = Wh·a_src, s_dst = Wh·a_dst
// grid = B*H*N blocks, 128 threads (one output row per block)
// ---------------------------------------------------------------------------
__global__ __launch_bounds__(FIN) void gat_k1_proj(
    const float* __restrict__ x, const float* __restrict__ W,
    const float* __restrict__ a, unsigned short* __restrict__ Whbf,
    float* __restrict__ ssrc, float* __restrict__ sdst)
{
    __shared__ float xrow[FIN];
    __shared__ float red[FOUT];
    int blk = blockIdx.x;
    int i = blk % Nn;
    int h = (blk / Nn) % Hh;
    int b = blk / (Nn * Hh);
    int f = threadIdx.x;

    xrow[f] = x[((size_t)b * Nn + i) * FIN + f];
    __syncthreads();

    const float* Wh_ = W + (size_t)h * FIN * FOUT;
    float acc = 0.f;
#pragma unroll 8
    for (int k = 0; k < FIN; ++k) acc = fmaf(xrow[k], Wh_[k * FOUT + f], acc);

    size_t bh = (size_t)b * Hh + h;
    Whbf[(bh * Nn + i) * FOUT + f] = f2bf(acc);

    red[f] = acc * a[h * 2 * FOUT + f];
    __syncthreads();
    for (int s = 64; s > 0; s >>= 1) { if (f < s) red[f] += red[f + s]; __syncthreads(); }
    if (f == 0) ssrc[bh * Nn + i] = red[0];
    __syncthreads();
    red[f] = acc * a[h * 2 * FOUT + FOUT + f];
    __syncthreads();
    for (int s = 64; s > 0; s >>= 1) { if (f < s) red[f] += red[f + s]; __syncthreads(); }
    if (f == 0) sdst[bh * Nn + i] = red[0];
}

// ---------------------------------------------------------------------------
// Kernel 2: denom[b,h,j] = sum_i |exp(leaky(ssrc_i + sdst_j)) * big_w[i,j]|
// only the 1024 rows where big_w is nonzero contribute.
// grid = B*H*(N/256) blocks, 256 threads (one column per thread)
// ---------------------------------------------------------------------------
__global__ __launch_bounds__(256) void gat_k2_denom(
    const int* __restrict__ adj, const float* __restrict__ wv,
    const float* __restrict__ ssrc, const float* __restrict__ sdst,
    float* __restrict__ denom)
{
    int blk = blockIdx.x;
    int jb = blk % (Nn / 256);
    int h  = (blk / (Nn / 256)) % Hh;
    int b  = blk / ((Nn / 256) * Hh);
    int j  = jb * 256 + threadIdx.x;
    int bh = b * Hh + h;

    float sd = sdst[bh * Nn + j];
    bool jtop = (j >= Uu);                 // right-half cols <- rows [0,U)
    int ibase = jtop ? 0 : Uu;             // left-half cols  <- rows [U,2U)
    const float* wvb = wv + (size_t)b * (Nn - Uu) * Uu;

    float acc = 0.f;
    for (int ii = 0; ii < Uu; ++ii) {
        int i = ibase + ii;
        float ss = ssrc[bh * Nn + i];
        float e = ss + sd; e = (e > 0.f) ? e : ALPHA * e;
        int msk = adj[((size_t)b * Nn + i) * Nn + j];
        float bw = jtop ? wvb[(size_t)(j - Uu) * Uu + ii]
                        : wvb[(size_t)ii * Uu + j];
        float v = (msk > 0) ? __expf(e) * bw : 0.f;
        acc += fabsf(v);
    }
    denom[bh * Nn + j] = acc;
}

// ---------------------------------------------------------------------------
// Kernel 3: out = elu(att @ Wh), fused: att tile computed on the fly in bf16,
// accumulated with v_wmma_f32_16x16x32_bf16. Block = 128x128 output tile,
// 8 waves x 16-row strips, K-loop of 32 over the 1024 nonzero columns.
// grid = B*H*(N/128), 256 threads
// ---------------------------------------------------------------------------
__global__ __launch_bounds__(256) void gat_k3_attn(
    const int* __restrict__ adj, const float* __restrict__ wv,
    const float* __restrict__ ssrc, const float* __restrict__ sdst,
    const float* __restrict__ denom, const unsigned short* __restrict__ Whbf,
    float* __restrict__ out)
{
    __shared__ unsigned short ldsA[TILE_M * APITCH];  // att tile, row-major [i][j]
    __shared__ unsigned short ldsB[FOUT * APITCH];    // Wh tile transposed [f][j]

    int blk = blockIdx.x;
    int ib = blk % (Nn / TILE_M);
    int h  = (blk / (Nn / TILE_M)) % Hh;
    int b  = blk / ((Nn / TILE_M) * Hh);
    int bh = b * Hh + h;
    int i0 = ib * TILE_M;
    bool top = (i0 < Uu);                 // top rows connect to cols [U,2U)
    int jbase = top ? Uu : 0;
    const float* wvb = wv + (size_t)b * (Nn - Uu) * Uu;

    int tid  = threadIdx.x;
    int lane = tid & 31;
    int wave = tid >> 5;
    int mrow = lane & 15;
    int half = lane >> 4;

    v8f acc[8] = {};

    for (int jt = 0; jt < Uu / KT; ++jt) {
        // ---- stage att tile (128 x 32) ----
        {
            int j = tid & 31;
            int irow0 = tid >> 5;
            int jg = jbase + jt * KT + j;
            float sd = sdst[bh * Nn + jg];
            float rden = 1.0f / fmaxf(denom[bh * Nn + jg], 1e-12f);
#pragma unroll
            for (int p = 0; p < 16; ++p) {
                int i = irow0 + p * 8;
                int ig = i0 + i;
                float ss = ssrc[bh * Nn + ig];
                float e = ss + sd; e = (e > 0.f) ? e : ALPHA * e;
                int msk = adj[((size_t)b * Nn + ig) * Nn + jg];
                float bw = top ? wvb[(size_t)(jg - Uu) * Uu + ig]
                               : wvb[(size_t)(ig - Uu) * Uu + jg];
                float v = (msk > 0) ? __expf(e) * bw * rden : 0.f;
                ldsA[i * APITCH + j] = f2bf(v);
            }
        }
        // ---- stage Wh tile (32 x 128), transposed into [f][j] ----
        {
            int f = tid & 127;
            int jr0 = tid >> 7;
#pragma unroll
            for (int p = 0; p < 16; ++p) {
                int jr = jr0 + p * 2;
                int jg = jbase + jt * KT + jr;
                ldsB[f * APITCH + jr] = Whbf[((size_t)bh * Nn + jg) * FOUT + f];
            }
        }
        __syncthreads();

        // ---- A fragment: 16x32 strip for this wave (ISA 16-bit A layout) ----
        v16bf afrag;
        {
            const unsigned short* pA = &ldsA[(wave * 16 + mrow) * APITCH + half * 8];
#pragma unroll
            for (int e = 0; e < 8; ++e) {
                afrag[e]     = bfc(pA[e]);        // K = kb .. kb+7
                afrag[8 + e] = bfc(pA[16 + e]);   // K = 16+kb .. 16+kb+7
            }
        }
        // ---- 8 B fragments (32x16 each) + WMMA ----
#pragma unroll
        for (int t = 0; t < 8; ++t) {
            v16bf bfrag;
            const unsigned short* pB = &ldsB[(t * 16 + mrow) * APITCH + half * 8];
#pragma unroll
            for (int e = 0; e < 16; ++e) bfrag[e] = bfc(pB[e < 8 ? e : 8 + e]);
            acc[t] = __builtin_amdgcn_wmma_f32_16x16x32_bf16(
                false, afrag, false, bfrag, (short)0, acc[t], false, false);
        }
        __syncthreads();
    }

    // ---- epilogue: ELU + store (C layout: VGPR r -> M = r + 8*half, N = mrow) ----
#pragma unroll
    for (int t = 0; t < 8; ++t) {
#pragma unroll
        for (int r = 0; r < 8; ++r) {
            int M = r + half * 8;
            int row = i0 + wave * 16 + M;
            int col = h * FOUT + t * 16 + mrow;
            float v = acc[t][r];
            v = (v > 0.f) ? v : (__expf(v) - 1.0f);
            out[((size_t)b * Nn + row) * (Hh * FOUT) + col] = v;
        }
    }
}

// ---------------------------------------------------------------------------
extern "C" void kernel_launch(void* const* d_in, const int* in_sizes, int n_in,
                              void* d_out, int out_size, void* d_ws, size_t ws_size,
                              hipStream_t stream)
{
    (void)in_sizes; (void)n_in; (void)out_size; (void)ws_size;
    const float* x   = (const float*)d_in[0];   // (B, N, FIN)
    const float* wv  = (const float*)d_in[1];   // (B, N-U, U)
    const float* W   = (const float*)d_in[2];   // (H, FIN, FOUT)
    const float* a   = (const float*)d_in[3];   // (H, 2*FOUT, 1)
    const int*   adj = (const int*)d_in[4];     // (B, N, N)
    float* out = (float*)d_out;                 // (B, N, H*FOUT)

    // workspace layout
    unsigned short* Whbf = (unsigned short*)d_ws;        // B*H*N*FOUT bf16 (4 MB)
    size_t whElems = (size_t)Bsz * Hh * Nn * FOUT;
    float* ssrc  = (float*)(Whbf + whElems);             // B*H*N f32
    float* sdst  = ssrc + (size_t)Bsz * Hh * Nn;
    float* denom = sdst + (size_t)Bsz * Hh * Nn;

    gat_k1_proj<<<Bsz * Hh * Nn, FIN, 0, stream>>>(x, W, a, Whbf, ssrc, sdst);
    gat_k2_denom<<<Bsz * Hh * (Nn / 256), 256, 0, stream>>>(adj, wv, ssrc, sdst, denom);
    gat_k3_attn<<<Bsz * Hh * (Nn / TILE_M), 256, 0, stream>>>(adj, wv, ssrc, sdst,
                                                              denom, Whbf, out);
}
